// MultiHeadSelfAttention_34187939676723
// MI455X (gfx1250) — matrix operand order repaired and checked by reading, not compiled, not run on hardware
//
#include <hip/hip_runtime.h>
#include <hip/hip_bf16.h>
#include <math.h>

// MHSA block: x -> QKV proj -> causal flash attention -> out proj + residual -> LayerNorm
// bf16 WMMA (v_wmma_f32_16x16x32_bf16) everywhere, fp32 accum.
// LDS tiles double-buffered and filled with GLOBAL_LOAD_ASYNC_TO_LDS_B128 (ASYNCcnt).
// Cross-lane reductions use ds_swizzle_b32 xor patterns (immediate, no index VGPR).

#define S_LEN  4096
#define D_DIM  1024
#define H_NUM  16
#define HD     64
#define LN_EPS 1e-5f

#ifndef USE_ASYNC_LDS
#define USE_ASYNC_LDS 1
#endif

typedef __attribute__((ext_vector_type(16))) __bf16        v16bf;
typedef __attribute__((ext_vector_type(8)))  float         v8f;
typedef __attribute__((ext_vector_type(8)))  unsigned int  v8u;

__device__ __forceinline__ unsigned short f2us(float f) {   // fp32 -> bf16 bits (RNE)
  union { float f; unsigned u; } x; x.f = f;
  unsigned r = x.u + 0x7FFFu + ((x.u >> 16) & 1u);
  return (unsigned short)(r >> 16);
}

__device__ __forceinline__ v8f wmma_bf16(v16bf a, v16bf b, v8f c) {
  return __builtin_amdgcn_wmma_f32_16x16x32_bf16(false, a, false, b, (short)0, c,
                                                 false, false);
}

// ds_swizzle xor-shuffle (group-of-32 mode: imm = xor<<10 | or<<5 | and)
#define SWZ_XOR(x, xorm)                                                     \
  __builtin_bit_cast(float, __builtin_amdgcn_ds_swizzle(                     \
      __builtin_bit_cast(int, (x)), (((xorm) << 10) | 0x1F)))

__device__ __forceinline__ float red_max16(float x) {   // max over 16-lane groups
  x = fmaxf(x, SWZ_XOR(x, 1));
  x = fmaxf(x, SWZ_XOR(x, 2));
  x = fmaxf(x, SWZ_XOR(x, 4));
  x = fmaxf(x, SWZ_XOR(x, 8));
  return x;
}
__device__ __forceinline__ float red_sum16(float x) {   // sum over 16-lane groups
  x += SWZ_XOR(x, 1);
  x += SWZ_XOR(x, 2);
  x += SWZ_XOR(x, 4);
  x += SWZ_XOR(x, 8);
  return x;
}
__device__ __forceinline__ float red_sum32(float x) {   // sum over full wave32
  x = red_sum16(x);
  x += SWZ_XOR(x, 16);
  return x;
}

// A-fragment: 16x32 bf16 (M=16, K=32), row-major, row stride ldk ushorts.
__device__ __forceinline__ v16bf frag_A(const unsigned short* base, int ldk, int lane) {
  const int m  = lane & 15;
  const int kb = (lane >> 4) << 3;           // 0 or 8
  const unsigned* p = (const unsigned*)(base + m * ldk);
  v8u u;
#pragma unroll
  for (int i = 0; i < 4; ++i) u[i]     = p[(kb >> 1) + i];
#pragma unroll
  for (int i = 0; i < 4; ++i) u[4 + i] = p[((16 + kb) >> 1) + i];
  return __builtin_bit_cast(v16bf, u);
}

// B-fragment: 32x16 bf16 (K=32, N=16), stored N-rows x K-contiguous, row stride ldk.
__device__ __forceinline__ v16bf frag_B(const unsigned short* base, int ldk, int lane) {
  const int n  = lane & 15;
  const int kb = (lane >> 4) << 3;
  const unsigned* p = (const unsigned*)(base + n * ldk);
  v8u u;
#pragma unroll
  for (int i = 0; i < 4; ++i) u[i]     = p[(kb >> 1) + i];
#pragma unroll
  for (int i = 0; i < 4; ++i) u[4 + i] = p[((16 + kb) >> 1) + i];
  return __builtin_bit_cast(v16bf, u);
}

// ---------------- async (ASYNCcnt) memory->LDS tile fill, 64 rows x 128 bytes ----------------
#if USE_ASYNC_LDS
__device__ __forceinline__ void async_load_b128(unsigned lds_off, unsigned long long gaddr) {
  asm volatile("global_load_async_to_lds_b128 %0, %1, off"
               :: "v"(lds_off), "v"(gaddr) : "memory");
}
__device__ __forceinline__ void wait_async_12() { asm volatile("s_wait_asynccnt 0xc" ::: "memory"); }
__device__ __forceinline__ void wait_async_8()  { asm volatile("s_wait_asynccnt 0x8" ::: "memory"); }
__device__ __forceinline__ void wait_async_0()  { asm volatile("s_wait_asynccnt 0x0" ::: "memory"); }

// Cooperative (128-thread) fill: 64 rows x 128B, LDS rows contiguous (128B stride).
// Per thread: 4 x b128 async issues. All addresses 16B aligned.
__device__ __forceinline__ void tile_fill_64x128(unsigned short* lds_tile,
                                                 const unsigned short* gbase,
                                                 int grow_ushorts, int t) {
  unsigned lbase = (unsigned)(uintptr_t)lds_tile;
  unsigned long long g = (unsigned long long)(uintptr_t)gbase;
#pragma unroll
  for (int i = 0; i < 4; ++i) {
    int c = t + i * 128;                      // 512 chunks of 16B
    int r = c >> 3, b = (c & 7) * 16;
    async_load_b128(lbase + (unsigned)(r * 128 + b),
                    g + (unsigned long long)r * ((unsigned)grow_ushorts * 2u) + (unsigned)b);
  }
}
#else
__device__ __forceinline__ void wait_async_12() {}
__device__ __forceinline__ void wait_async_8()  {}
__device__ __forceinline__ void wait_async_0()  {}
__device__ __forceinline__ void tile_fill_64x128(unsigned short* lds_tile,
                                                 const unsigned short* gbase,
                                                 int grow_ushorts, int t) {
  unsigned* l = (unsigned*)lds_tile;
  const unsigned* g = (const unsigned*)gbase;
  int grow_u = grow_ushorts >> 1;
#pragma unroll
  for (int i = 0; i < 16; ++i) {              // 2048 uints / 128 threads
    int c = t + i * 128;
    int r = c >> 5, j = c & 31;
    l[c] = g[r * grow_u + j];
  }
}
#endif

// ---------------------------------------------------------------- fp32 -> bf16 convert
__global__ void cvt_bf16_kernel(const float4* __restrict__ in, uint2* __restrict__ out, int n4) {
  int i = blockIdx.x * 256 + threadIdx.x;
  if (i < n4) {
    float4 f = in[i];
    uint2 o;
    o.x = (unsigned)f2us(f.x) | ((unsigned)f2us(f.y) << 16);
    o.y = (unsigned)f2us(f.z) | ((unsigned)f2us(f.w) << 16);
    out[i] = o;
  }
}

// ---------------------------------------------------------------- WMMA GEMM (x @ W^T + b)
// WG = 128 rows x 64 cols, 4 waves, 32 rows/wave (16 WMMAs per k-step per wave).
// MODE 0: bf16 out[s*D + n];  MODE 1: bf16 out[n*S + s] (V^T);  MODE 2: f32 out + residual.
template <int MODE>
__global__ __launch_bounds__(128) void gemm_bf16_kernel(
    const unsigned short* __restrict__ A,      // [S_LEN][D_DIM] bf16
    const unsigned short* __restrict__ W,      // [D_DIM][D_DIM] bf16 row-major
    const float* __restrict__ bias,
    const float* __restrict__ residual,        // f32 or nullptr
    void* __restrict__ out,
    float scale) {
  __shared__ unsigned short Xt[2][128 * 64];   // 128 s-rows x 64 k (16KB/buf)
  __shared__ unsigned short Wt[2][64 * 64];    // 64 n-rows x 64 k  (8KB/buf)
  const int t    = threadIdx.x;
  const int lane = t & 31;
  const int w    = t >> 5;
  const int s0   = blockIdx.x * 128;
  const int n0   = blockIdx.y * 64;

  const unsigned short* Ag = A + (size_t)s0 * D_DIM;
  const unsigned short* Wg = W + (size_t)n0 * D_DIM;

  v8f acc[2][4] = {{v8f{}, v8f{}, v8f{}, v8f{}}, {v8f{}, v8f{}, v8f{}, v8f{}}};

  // preload k-slab 0: 12 async issues/thread (8 for X, 4 for W)
  tile_fill_64x128(Xt[0],           Ag,               D_DIM, t);
  tile_fill_64x128(Xt[0] + 64 * 64, Ag + 64 * D_DIM,  D_DIM, t);
  tile_fill_64x128(Wt[0],           Wg,               D_DIM, t);

  const int NIT = D_DIM / 64;                  // 16
  for (int it = 0; it < NIT; ++it) {
    const int cur = it & 1;
    if (it + 1 < NIT) {
      const unsigned short* Agn = Ag + (it + 1) * 64;
      tile_fill_64x128(Xt[cur ^ 1],           Agn,              D_DIM, t);
      tile_fill_64x128(Xt[cur ^ 1] + 64 * 64, Agn + 64 * D_DIM, D_DIM, t);
      tile_fill_64x128(Wt[cur ^ 1],           Wg + (it + 1) * 64, D_DIM, t);
      wait_async_12();                         // current slab landed (in-order), next in flight
    } else {
      wait_async_0();
    }
    __syncthreads();                           // all waves' fills visible

    const unsigned short* xa = Xt[cur] + (w * 32) * 64;
    v16bf a00 = frag_A(xa,            64, lane);
    v16bf a01 = frag_A(xa + 32,       64, lane);
    v16bf a10 = frag_A(xa + 16 * 64,      64, lane);
    v16bf a11 = frag_A(xa + 16 * 64 + 32, 64, lane);
#pragma unroll
    for (int nt = 0; nt < 4; ++nt) {
      const unsigned short* wb = Wt[cur] + nt * 16 * 64;
      v16bf b0 = frag_B(wb,      64, lane);
      v16bf b1 = frag_B(wb + 32, 64, lane);
      acc[0][nt] = wmma_bf16(a00, b0, acc[0][nt]);
      acc[0][nt] = wmma_bf16(a01, b1, acc[0][nt]);
      acc[1][nt] = wmma_bf16(a10, b0, acc[1][nt]);
      acc[1][nt] = wmma_bf16(a11, b1, acc[1][nt]);
    }
    __syncthreads();                           // done reading cur before it is refilled
  }

  const int mrow = (lane >> 4) << 3;
  const int n_l  = lane & 15;
#pragma unroll
  for (int sub = 0; sub < 2; ++sub) {
#pragma unroll
    for (int nt = 0; nt < 4; ++nt) {
#pragma unroll
      for (int r = 0; r < 8; ++r) {
        int s = s0 + w * 32 + sub * 16 + mrow + r;
        int n = n0 + nt * 16 + n_l;
        float v = (acc[sub][nt][r] + bias[n]) * scale;
        if constexpr (MODE == 0) {
          ((unsigned short*)out)[(size_t)s * D_DIM + n] = f2us(v);
        } else if constexpr (MODE == 1) {
          ((unsigned short*)out)[(size_t)n * S_LEN + s] = f2us(v);
        } else {
          ((float*)out)[(size_t)s * D_DIM + n] = v + residual[(size_t)s * D_DIM + n];
        }
      }
    }
  }
}

// ---------------------------------------------------------------- causal flash attention
// Q,K: [S][D] bf16 (Q pre-scaled 1/8); Vt: [D][S] bf16; Oa: [S][D] bf16.
// WG = 64 query rows x 1 head, 4 waves (16 rows each), key blocks of 64, double-buffered.
__global__ __launch_bounds__(128) void attn_kernel(
    const unsigned short* __restrict__ Q,
    const unsigned short* __restrict__ K,
    const unsigned short* __restrict__ Vt,
    unsigned short* __restrict__ Oa) {
  __shared__ unsigned short Kt[2][64 * 64];    // 64 keys x 64 d
  __shared__ unsigned short Vl[2][64 * 64];    // 64 d x 64 keys
  __shared__ unsigned short Pt[4][16 * 64];    // per-wave probs (A-frag staging)

  const int t    = threadIdx.x;
  const int lane = t & 31;
  const int w    = t >> 5;
  const int q0   = blockIdx.x * 64;
  const int h    = blockIdx.y;
  const int qrow = q0 + w * 16;
  const int mrow = (lane >> 4) << 3;
  const int n_l  = lane & 15;
  const float NEG_INF = -__builtin_inff();

  v16bf qa0 = frag_A(Q + (size_t)qrow * D_DIM + h * HD,      D_DIM, lane);
  v16bf qa1 = frag_A(Q + (size_t)qrow * D_DIM + h * HD + 32, D_DIM, lane);

  v8f acc[4] = {v8f{}, v8f{}, v8f{}, v8f{}};
  float mrun[8], lrun[8];
#pragma unroll
  for (int r = 0; r < 8; ++r) { mrun[r] = NEG_INF; lrun[r] = 0.f; }

  const unsigned short* Kg = K + h * HD;                     // + kb*D_DIM per block
  const unsigned short* Vg = Vt + (size_t)h * HD * S_LEN;    // + kb per block
  const int nb = blockIdx.x + 1;                             // 64-key blocks (causal)

  tile_fill_64x128(Kt[0], Kg, D_DIM, t);
  tile_fill_64x128(Vl[0], Vg, S_LEN, t);

  for (int ib = 0; ib < nb; ++ib) {
    const int kb  = ib * 64;
    const int cur = ib & 1;
    if (ib + 1 < nb) {
      tile_fill_64x128(Kt[cur ^ 1], Kg + (size_t)(kb + 64) * D_DIM, D_DIM, t);
      tile_fill_64x128(Vl[cur ^ 1], Vg + (kb + 64),                 S_LEN, t);
      wait_async_8();
    } else {
      wait_async_0();
    }
    __syncthreads();

    // scores: 4 N-tiles, 2 k-chunks each
    v8f s[4] = {v8f{}, v8f{}, v8f{}, v8f{}};
#pragma unroll
    for (int nt = 0; nt < 4; ++nt) {
      const unsigned short* kbp = Kt[cur] + nt * 16 * 64;
      s[nt] = wmma_bf16(qa0, frag_B(kbp,      64, lane), s[nt]);
      s[nt] = wmma_bf16(qa1, frag_B(kbp + 32, 64, lane), s[nt]);
    }

    // streaming softmax per C-row
#pragma unroll
    for (int r = 0; r < 8; ++r) {
      int q = qrow + mrow + r;
      float v[4];
#pragma unroll
      for (int nt = 0; nt < 4; ++nt) {
        v[nt] = s[nt][r];
        if (kb + nt * 16 + n_l > q) v[nt] = NEG_INF;         // causal mask
      }
      float rm = red_max16(fmaxf(fmaxf(v[0], v[1]), fmaxf(v[2], v[3])));
      float nm = fmaxf(mrun[r], rm);
      float corr, p[4];
      if (nm == NEG_INF) {
        corr = 0.f;
#pragma unroll
        for (int nt = 0; nt < 4; ++nt) p[nt] = 0.f;
      } else {
        corr = (mrun[r] == NEG_INF) ? 0.f : __expf(mrun[r] - nm);
#pragma unroll
        for (int nt = 0; nt < 4; ++nt)
          p[nt] = (v[nt] == NEG_INF) ? 0.f : __expf(v[nt] - nm);
      }
      mrun[r] = nm;
      float rs = red_sum16((p[0] + p[1]) + (p[2] + p[3]));
      lrun[r] = lrun[r] * corr + rs;
#pragma unroll
      for (int dt = 0; dt < 4; ++dt) acc[dt][r] *= corr;
#pragma unroll
      for (int nt = 0; nt < 4; ++nt)
        Pt[w][(mrow + r) * 64 + nt * 16 + n_l] = f2us(p[nt]);
    }
    asm volatile("s_wait_dscnt 0" ::: "memory");             // own-wave LDS RAW
    v16bf pa0 = frag_A(&Pt[w][0],      64, lane);
    v16bf pa1 = frag_A(&Pt[w][0] + 32, 64, lane);
#pragma unroll
    for (int dt = 0; dt < 4; ++dt) {
      const unsigned short* vb = Vl[cur] + dt * 16 * 64;
      acc[dt] = wmma_bf16(pa0, frag_B(vb,      64, lane), acc[dt]);
      acc[dt] = wmma_bf16(pa1, frag_B(vb + 32, 64, lane), acc[dt]);
    }
    __syncthreads();                                         // done with cur before refill
  }

#pragma unroll
  for (int dt = 0; dt < 4; ++dt) {
#pragma unroll
    for (int r = 0; r < 8; ++r) {
      int q = qrow + mrow + r;
      float inv = 1.0f / lrun[r];
      Oa[(size_t)q * D_DIM + h * HD + dt * 16 + n_l] = f2us(acc[dt][r] * inv);
    }
  }
}

// ---------------------------------------------------------------- LayerNorm (one row / WG)
__global__ __launch_bounds__(256) void ln_kernel(const float* __restrict__ Y,
                                                 const float* __restrict__ gamma,
                                                 const float* __restrict__ beta,
                                                 float* __restrict__ out) {
  __shared__ float red[16];
  const int row = blockIdx.x;
  const int t   = threadIdx.x;
  const float* y = Y + (size_t)row * D_DIM;
  float v[4], s = 0.f, ss = 0.f;
#pragma unroll
  for (int i = 0; i < 4; ++i) {
    v[i] = y[t + i * 256];
    s  += v[i];
    ss += v[i] * v[i];
  }
  s  = red_sum32(s);
  ss = red_sum32(ss);
  if ((t & 31) == 0) { red[t >> 5] = s; red[8 + (t >> 5)] = ss; }
  __syncthreads();
  float ts = 0.f, tss = 0.f;
#pragma unroll
  for (int i = 0; i < 8; ++i) { ts += red[i]; tss += red[8 + i]; }
  float mean = ts * (1.0f / D_DIM);
  float var  = tss * (1.0f / D_DIM) - mean * mean;
  float rstd = rsqrtf(var + LN_EPS);
#pragma unroll
  for (int i = 0; i < 4; ++i) {
    int c = t + i * 256;
    out[(size_t)row * D_DIM + c] = (v[i] - mean) * rstd * gamma[c] + beta[c];
  }
}

// ---------------------------------------------------------------- host launcher
extern "C" void kernel_launch(void* const* d_in, const int* in_sizes, int n_in,
                              void* d_out, int out_size, void* d_ws, size_t ws_size,
                              hipStream_t stream) {
  (void)in_sizes; (void)n_in; (void)out_size; (void)ws_size;
  const float* input = (const float*)d_in[0];
  const float* Wq = (const float*)d_in[1];
  const float* bq = (const float*)d_in[2];
  const float* Wk = (const float*)d_in[3];
  const float* bk = (const float*)d_in[4];
  const float* Wv = (const float*)d_in[5];
  const float* bv = (const float*)d_in[6];
  const float* Wo = (const float*)d_in[7];
  const float* bo = (const float*)d_in[8];
  const float* g  = (const float*)d_in[9];
  const float* bt = (const float*)d_in[10];

  char* ws = (char*)d_ws;
  unsigned short* Xb  = (unsigned short*)(ws);                         //  0MB: 8MB
  unsigned short* Wqb = (unsigned short*)(ws + (size_t)( 8 << 20));    //  8MB: 2MB
  unsigned short* Wkb = Wqb + (size_t)D_DIM * D_DIM;                   // 10MB
  unsigned short* Wvb = Wkb + (size_t)D_DIM * D_DIM;                   // 12MB
  unsigned short* Wob = Wvb + (size_t)D_DIM * D_DIM;                   // 14MB
  unsigned short* Qb  = (unsigned short*)(ws + (size_t)(16 << 20));    // 16MB: 8MB
  unsigned short* Kb  = (unsigned short*)(ws + (size_t)(24 << 20));    // 24MB: 8MB
  unsigned short* Vt  = (unsigned short*)(ws + (size_t)(32 << 20));    // 32MB: 8MB
  unsigned short* Ab  = (unsigned short*)(ws + (size_t)(40 << 20));    // 40MB: 8MB
  float*          Yf  = (float*)(ws + (size_t)(16 << 20));             // reuse Q/K: 16MB

  const int nSD4 = (S_LEN * D_DIM) / 4;
  const int nDD4 = (D_DIM * D_DIM) / 4;
  cvt_bf16_kernel<<<nSD4 / 256, 256, 0, stream>>>((const float4*)input, (uint2*)Xb,  nSD4);
  cvt_bf16_kernel<<<nDD4 / 256, 256, 0, stream>>>((const float4*)Wq,    (uint2*)Wqb, nDD4);
  cvt_bf16_kernel<<<nDD4 / 256, 256, 0, stream>>>((const float4*)Wk,    (uint2*)Wkb, nDD4);
  cvt_bf16_kernel<<<nDD4 / 256, 256, 0, stream>>>((const float4*)Wv,    (uint2*)Wvb, nDD4);
  cvt_bf16_kernel<<<nDD4 / 256, 256, 0, stream>>>((const float4*)Wo,    (uint2*)Wob, nDD4);

  dim3 gg(S_LEN / 128, D_DIM / 64);
  const float qscale = 0.125f;   // 1/sqrt(Hd=64)
  gemm_bf16_kernel<0><<<gg, 128, 0, stream>>>(Xb, Wqb, bq, nullptr, Qb, qscale);
  gemm_bf16_kernel<0><<<gg, 128, 0, stream>>>(Xb, Wkb, bk, nullptr, Kb, 1.0f);
  gemm_bf16_kernel<1><<<gg, 128, 0, stream>>>(Xb, Wvb, bv, nullptr, Vt, 1.0f);

  dim3 ga(S_LEN / 64, H_NUM);
  attn_kernel<<<ga, 128, 0, stream>>>(Qb, Kb, Vt, Ab);

  gemm_bf16_kernel<2><<<gg, 128, 0, stream>>>(Ab, Wob, bo, input, Yf, 1.0f);
  ln_kernel<<<S_LEN, 256, 0, stream>>>(Yf, g, bt, (float*)d_out);
}
